// RelPositionMultiHeadAttention_59133109731776
// MI455X (gfx1250) — compile-verified, matrix operand order checked by reading, not run on hardware
//
#include <hip/hip_runtime.h>

typedef __attribute__((ext_vector_type(16))) __bf16 v16bf;
typedef __attribute__((ext_vector_type(8)))  __bf16 v8bf;
typedef __attribute__((ext_vector_type(4)))  __bf16 v4bf;
typedef __attribute__((ext_vector_type(8)))  float  v8f;
typedef __attribute__((ext_vector_type(4)))  float  v4f;
typedef __attribute__((ext_vector_type(4)))  int    v4i;

#define T_LEN 1024
#define FEATN 1024
#define DH    64
#define PLEN  2047
#define INF_VAL 10000.0f

#if defined(__AMDGCN__) && __has_builtin(__builtin_amdgcn_global_load_async_to_lds_b128) && \
    __has_builtin(__builtin_amdgcn_s_wait_asynccnt)
#define USE_ASYNC 1
#else
#define USE_ASYNC 0
#endif

// 16-byte global -> LDS copy. Async (ASYNCcnt-tracked, no VGPR round trip)
// when the gfx1250 builtin is available; plain copy otherwise.
static __device__ __forceinline__ void cp16_async(void* l, const void* g) {
#if USE_ASYNC
  __builtin_amdgcn_global_load_async_to_lds_b128(
      (__attribute__((address_space(1))) v4i*)g,
      (__attribute__((address_space(3))) v4i*)l, 0, 0);
#else
  *(v4f*)l = *(const v4f*)g;
#endif
}
static __device__ __forceinline__ void async_fence() {
#if USE_ASYNC
  __builtin_amdgcn_s_wait_asynccnt(0);
#endif
}

static __device__ __forceinline__ v8f wmma_bf16(v16bf a, v16bf b, v8f c) {
  return __builtin_amdgcn_wmma_f32_16x16x32_bf16(false, a, false, b, (short)0, c, false, false);
}

// WMMA fragment for lane (r16, half) from a K-contiguous bf16 row:
// two ds_load_b128 (elements k = half*8+0..7 and 16+half*8+0..7).
static __device__ __forceinline__ v16bf frag16(const __bf16* rowp, int half) {
  v8bf lo = *(const v8bf*)(rowp + half * 8);
  v8bf hi = *(const v8bf*)(rowp + 16 + half * 8);
  return __builtin_shufflevector(lo, hi, 0,1,2,3,4,5,6,7,8,9,10,11,12,13,14,15);
}

// Same from an f32 row (4x ds_load_b128 + packed converts).
static __device__ __forceinline__ v16bf frag16_f32(const float* rowp, int half) {
  v4f a = *(const v4f*)(rowp + half * 8);
  v4f b = *(const v4f*)(rowp + half * 8 + 4);
  v4f c = *(const v4f*)(rowp + 16 + half * 8);
  v4f d = *(const v4f*)(rowp + 16 + half * 8 + 4);
  v16bf r;
#pragma unroll
  for (int q = 0; q < 4; ++q) {
    r[q]      = (__bf16)a[q];
    r[4 + q]  = (__bf16)b[q];
    r[8 + q]  = (__bf16)c[q];
    r[12 + q] = (__bf16)d[q];
  }
  return r;
}

// ---------------------------------------------------------------------------
// GEMM: C = A(MxK,f32) @ W(KxN,f32). Block 128x64, K-step 32, double-buffered.
// A tile staged raw f32 via async-to-LDS (converted at fragment build);
// W tile staged transposed [n][k] bf16 (needs convert+transpose -> sync path).
// MODE 0: out1 = C (+bias0) -> bf16 | 1: dual bf16 +extras | 2: outf f32
// ---------------------------------------------------------------------------
template <int MODE, bool FULL_M>
__global__ __launch_bounds__(256)
void gemm_bias_kernel(const float* __restrict__ A, const float* __restrict__ W,
                      const float* __restrict__ bias0,
                      const float* __restrict__ extra1, const float* __restrict__ extra2,
                      __bf16* __restrict__ out1, __bf16* __restrict__ out2,
                      float* __restrict__ outf,
                      int M, int N, int K)
{
  __shared__ __align__(16) float  Asf[2][128][32];  // [buf][m][k] raw f32
  __shared__ __align__(16) __bf16 Bt[2][64][40];    // [buf][n][k] bf16

  const int tid  = threadIdx.x;
  const int wave = tid >> 5, lane = tid & 31;
  const int r16  = lane & 15, half = lane >> 4;
  const int wm   = wave & 3;
  const int wn   = wave >> 2;
  const int m0   = blockIdx.x * 128;
  const int n0   = blockIdx.y * 64;

  auto stageA = [&](int buf, int k0) {
#pragma unroll
    for (int i = tid; i < 1024; i += 256) {          // 4 async b128 per thread
      int r = i >> 3, c4 = (i & 7) * 4;
      int row = m0 + r;
      if (!FULL_M) row = (row < M) ? row : (M - 1);  // clamp: edge rows unused at store
      cp16_async(&Asf[buf][r][c4], &A[(size_t)row * K + k0 + c4]);
    }
  };
  auto stageW = [&](int buf, int k0) {
#pragma unroll
    for (int i = tid; i < 512; i += 256) {
      int r = i >> 4, c4 = (i & 15) * 4;
      v4f w = *(const v4f*)&W[(size_t)(k0 + r) * N + n0 + c4];
#pragma unroll
      for (int q = 0; q < 4; ++q) Bt[buf][c4 + q][r] = (__bf16)w[q];
    }
  };

  const v8f vzero = {0.f,0.f,0.f,0.f,0.f,0.f,0.f,0.f};
  v8f acc[2][2];
#pragma unroll
  for (int i = 0; i < 2; ++i)
#pragma unroll
    for (int j = 0; j < 2; ++j) acc[i][j] = vzero;

  const int NK = K >> 5;
  stageA(0, 0); stageW(0, 0);
  for (int ks = 0; ks < NK; ++ks) {
    async_fence();
    __syncthreads();
    if (ks + 1 < NK) { stageA((ks + 1) & 1, (ks + 1) << 5);
                       stageW((ks + 1) & 1, (ks + 1) << 5); }
    const int cur = ks & 1;
    v16bf a0 = frag16_f32(&Asf[cur][wm * 32 +      r16][0], half);
    v16bf a1 = frag16_f32(&Asf[cur][wm * 32 + 16 + r16][0], half);
    v16bf b0 = frag16(&Bt[cur][wn * 32 +      r16][0], half);
    v16bf b1 = frag16(&Bt[cur][wn * 32 + 16 + r16][0], half);
    acc[0][0] = wmma_bf16(a0, b0, acc[0][0]);
    acc[0][1] = wmma_bf16(a0, b1, acc[0][1]);
    acc[1][0] = wmma_bf16(a1, b0, acc[1][0]);
    acc[1][1] = wmma_bf16(a1, b1, acc[1][1]);
  }

  float b0c[2] = {0.f, 0.f}, e1c[2] = {0.f, 0.f}, e2c[2] = {0.f, 0.f};
#pragma unroll
  for (int ni = 0; ni < 2; ++ni) {
    int col = n0 + wn * 32 + ni * 16 + r16;
    if (bias0) b0c[ni] = bias0[col];
    if (MODE == 1) { e1c[ni] = extra1[col]; e2c[ni] = extra2[col]; }
  }
#pragma unroll
  for (int mi = 0; mi < 2; ++mi)
#pragma unroll
    for (int ni = 0; ni < 2; ++ni)
#pragma unroll
      for (int v = 0; v < 8; ++v) {
        int row = m0 + wm * 32 + mi * 16 + half * 8 + v;
        int col = n0 + wn * 32 + ni * 16 + r16;
        if (FULL_M || row < M) {
          float base = acc[mi][ni][v] + b0c[ni];
          size_t o = (size_t)row * N + col;
          if (MODE == 0) out1[o] = (__bf16)base;
          if (MODE == 1) { out1[o] = (__bf16)(base + e1c[ni]);
                           out2[o] = (__bf16)(base + e2c[ni]); }
          if (MODE == 2) outf[o] = base;
        }
      }
}

// ---------------------------------------------------------------------------
// Fused rel-pos attention per (b, h, 32-row query block).
// rel_shift identity: shifted_bd[i,j] = q_v[i] . p[T-1-i+j]
// ---------------------------------------------------------------------------
__global__ __launch_bounds__(256)
void attention_kernel(const __bf16* __restrict__ qu, const __bf16* __restrict__ qv,
                      const __bf16* __restrict__ kb, const __bf16* __restrict__ vb,
                      const __bf16* __restrict__ pb, const unsigned char* __restrict__ mask,
                      float* __restrict__ xout)
{
  extern __shared__ __align__(16) char smem[];
  float*  S    = (float*)smem;                       // [32][1024] f32 (128 KB)
  __bf16* squ  = (__bf16*)(smem + 131072);           // [32][64]
  __bf16* sqv  = squ + 32 * 64;                      // [32][64]
  __bf16* ktl  = sqv + 32 * 64;                      // [2][128][64]
  __bf16* pt   = ktl + 2 * 128 * 64;                 // [2][96][64]
  float*  bdt  = (float*)(pt + 2 * 96 * 64);         // [32][96] f32
  __bf16* vtT  = (__bf16*)(bdt + 32 * 96);           // [2][64][32] (V^T: [d][k])

  const int tid  = threadIdx.x;
  const int wave = tid >> 5, lane = tid & 31;
  const int r16  = lane & 15, half = lane >> 4;
  const int i0   = blockIdx.x * 32;
  const int h    = blockIdx.y;
  const int b    = blockIdx.z;
  const int mi   = wave >> 2;   // 0..1: 16-row group
  const int sub  = wave & 3;    // 0..3
  const v8f vzero = {0.f,0.f,0.f,0.f,0.f,0.f,0.f,0.f};

  auto stageK = [&](int buf, int j0) {               // 128 key rows, 4 async/thread
#pragma unroll
    for (int i = tid; i < 1024; i += 256) {
      int r = i >> 3, c8 = (i & 7) * 8;
      cp16_async(&ktl[buf * 8192 + r * 64 + c8],
                 &kb[((size_t)(b * T_LEN + j0 + r) * FEATN) + h * DH + c8]);
    }
  };
  auto stageP = [&](int buf, int rel0) {
#pragma unroll
    for (int i = tid; i < 768; i += 256) {
      int r = i >> 3, c8 = (i & 7) * 8;
      int prow = rel0 + r;
      prow = (prow < PLEN) ? prow : (PLEN - 1);      // clamp: feeds only unread bdt col 95
      cp16_async(&pt[buf * 6144 + r * 64 + c8],
                 &pb[(size_t)prow * FEATN + h * DH + c8]);
    }
  };
  auto stageV = [&](int buf, int kk) {               // transpose -> sync path
    for (int i = tid; i < 256; i += 256) {
      int r = i >> 3, c8 = (i & 7) * 8;
      v8bf val = *(const v8bf*)&vb[((size_t)(b * T_LEN + kk + r) * FEATN) + h * DH + c8];
#pragma unroll
      for (int q = 0; q < 8; ++q) vtT[buf * 2048 + (c8 + q) * 32 + r] = val[q];
    }
  };

  // Stage q_u / q_v + first K tile
  for (int i = tid; i < 256; i += 256) {
    int r = i >> 3, c8 = (i & 7) * 8;
    size_t g = ((size_t)(b * T_LEN + i0 + r) * FEATN) + h * DH + c8;
    cp16_async(&squ[r * 64 + c8], &qu[g]);
    cp16_async(&sqv[r * 64 + c8], &qv[g]);
  }
  stageK(0, 0);
  async_fence();
  __syncthreads();

  // ---- Phase 1: AC = q_u @ k^T; 128-col chunks; 2 independent accs/wave ----
  {
    v16bf a0 = frag16(&squ[(mi * 16 + r16) * 64], half);
    v16bf a1 = frag16(&squ[(mi * 16 + r16) * 64 + 32], half);
    for (int jt = 0; jt < 8; ++jt) {
      if (jt + 1 < 8) stageK((jt + 1) & 1, (jt + 1) * 128);
      const __bf16* kcur = ktl + (jt & 1) * 8192;
      v16bf b00 = frag16(&kcur[( sub      * 16 + r16) * 64], half);
      v16bf b01 = frag16(&kcur[( sub      * 16 + r16) * 64 + 32], half);
      v16bf b10 = frag16(&kcur[((sub + 4) * 16 + r16) * 64], half);
      v16bf b11 = frag16(&kcur[((sub + 4) * 16 + r16) * 64 + 32], half);
      v8f acc0 = vzero, acc1 = vzero;
      acc0 = wmma_bf16(a0, b00, acc0);
      acc1 = wmma_bf16(a0, b10, acc1);
      acc0 = wmma_bf16(a1, b01, acc0);
      acc1 = wmma_bf16(a1, b11, acc1);
#pragma unroll
      for (int v = 0; v < 8; ++v) {
        int srow = (mi * 16 + half * 8 + v) * 1024 + jt * 128;
        S[srow + sub * 16 + r16]      = acc0[v];
        S[srow + sub * 16 + 64 + r16] = acc1[v];
      }
      async_fence();
      __syncthreads();
    }
  }

  // ---- Phase 2: BD band WMMA + diagonal gather + scale/mask ----
  {
    v16bf a0 = frag16(&sqv[(mi * 16 + r16) * 64], half);
    v16bf a1 = frag16(&sqv[(mi * 16 + r16) * 64 + 32], half);
    stageP(0, T_LEN - 32 - i0);
    for (int jc = 0; jc < 16; ++jc) {
      async_fence();
      __syncthreads();
      const int j0 = jc * 64;
      if (jc + 1 < 16) stageP((jc + 1) & 1, (jc + 1) * 64 + T_LEN - 32 - i0);
      const __bf16* pcur = pt + (jc & 1) * 6144;
      if (sub < 3) {                      // waves sub 0..2 own tile pair (2sub, 2sub+1)
        const int tnA = sub * 2, tnB = sub * 2 + 1;
        v16bf b00 = frag16(&pcur[(tnA * 16 + r16) * 64], half);
        v16bf b01 = frag16(&pcur[(tnA * 16 + r16) * 64 + 32], half);
        v16bf b10 = frag16(&pcur[(tnB * 16 + r16) * 64], half);
        v16bf b11 = frag16(&pcur[(tnB * 16 + r16) * 64 + 32], half);
        v8f acc0 = vzero, acc1 = vzero;
        acc0 = wmma_bf16(a0, b00, acc0);
        acc1 = wmma_bf16(a0, b10, acc1);
        acc0 = wmma_bf16(a1, b01, acc0);
        acc1 = wmma_bf16(a1, b11, acc1);
#pragma unroll
        for (int v = 0; v < 8; ++v) {
          int brow = (mi * 16 + half * 8 + v) * 96;
          bdt[brow + tnA * 16 + r16] = acc0[v];
          bdt[brow + tnB * 16 + r16] = acc1[v];
        }
      }
      __syncthreads();
      for (int i = tid; i < 32 * 64; i += 256) {
        int r = i >> 6, c = i & 63;
        int rr = c + 31 - r;                         // rel_shift band offset (<= 94)
        float sc = (S[r * 1024 + j0 + c] + bdt[r * 96 + rr]) * 0.125f; // 1/sqrt(64)
        unsigned char m = mask[((size_t)(b * T_LEN + i0 + r)) * T_LEN + j0 + c];
        S[r * 1024 + j0 + c] = m ? -INF_VAL : sc;
      }
    }
    __syncthreads();
  }

  // ---- Phase 3: softmax (each wave owns 4 rows) ----
  for (int rr = wave * 4; rr < wave * 4 + 4; ++rr) {
    float* row = S + rr * 1024;
    float mx = -3.0e38f;
    for (int j = lane; j < 1024; j += 32) mx = fmaxf(mx, row[j]);
#pragma unroll
    for (int off = 16; off > 0; off >>= 1) mx = fmaxf(mx, __shfl_xor(mx, off, 32));
    float sum = 0.0f;
    for (int j = lane; j < 1024; j += 32) {
      float e = __expf(row[j] - mx);
      row[j] = e;
      sum += e;
    }
#pragma unroll
    for (int off = 16; off > 0; off >>= 1) sum += __shfl_xor(sum, off, 32);
    float inv = 1.0f / sum;
    const unsigned char* mrow = mask + ((size_t)(b * T_LEN + i0 + rr)) * T_LEN;
    for (int j = lane; j < 1024; j += 32)
      row[j] = mrow[j] ? 0.0f : row[j] * inv;
  }

  // ---- Phase 4: O = attn @ V (vtT double-buffered) ----
  stageV(0, 0);
  __syncthreads();
  const int nd = wave & 3;
  v8f oacc = vzero;
  for (int ks = 0; ks < 32; ++ks) {
    if (ks + 1 < 32) stageV((ks + 1) & 1, (ks + 1) * 32);
    const __bf16* vcur = vtT + (ks & 1) * 2048;
    v16bf a  = frag16_f32(&S[(mi * 16 + r16) * 1024 + ks * 32], half);
    v16bf bf = frag16(&vcur[(nd * 16 + r16) * 32], half);
    oacc = wmma_bf16(a, bf, oacc);
    __syncthreads();
  }
#pragma unroll
  for (int v = 0; v < 8; ++v) {
    int row = i0 + mi * 16 + half * 8 + v;
    xout[((size_t)(b * T_LEN + row) * FEATN) + h * DH + nd * 16 + r16] = oacc[v];
  }
}

// ---------------------------------------------------------------------------
extern "C" void kernel_launch(void* const* d_in, const int* in_sizes, int n_in,
                              void* d_out, int out_size, void* d_ws, size_t ws_size,
                              hipStream_t stream)
{
  const float* query   = (const float*)d_in[0];
  const float* key_    = (const float*)d_in[1];
  const float* value   = (const float*)d_in[2];
  const unsigned char* mask = (const unsigned char*)d_in[3];  // bool, all-false
  const float* pos_emb = (const float*)d_in[4];
  const float* Wq   = (const float*)d_in[5];
  const float* bq   = (const float*)d_in[6];
  const float* Wk   = (const float*)d_in[7];
  const float* bk   = (const float*)d_in[8];
  const float* Wv   = (const float*)d_in[9];
  const float* bv   = (const float*)d_in[10];
  const float* Wpos = (const float*)d_in[11];
  const float* pbu  = (const float*)d_in[12];   // [H,D] == per-feature bias
  const float* pbv  = (const float*)d_in[13];
  const float* Wout = (const float*)d_in[14];
  const float* bout = (const float*)d_in[15];

  char* ws = (char*)d_ws;
  __bf16* quB = (__bf16*)(ws + 0);          // q + bias_u  [B*T, FEAT]
  __bf16* qvB = (__bf16*)(ws + 8388608);    // q + bias_v
  __bf16* kB  = (__bf16*)(ws + 16777216);
  __bf16* vB  = (__bf16*)(ws + 25165824);
  __bf16* pB  = (__bf16*)(ws + 33554432);   // [P, FEAT]
  float*  x   = (float*)(ws + 37748736);    // [B*T, FEAT]

  dim3 blk(256);
  dim3 gMain(32, 16);   // M=4096/128, N=1024/64
  dim3 gPos(16, 16);    // M=2047 -> 16 tiles (clamp + store guard)

  gemm_bias_kernel<1, true ><<<gMain, blk, 0, stream>>>(query, Wq, bq, pbu, pbv,
                                              quB, qvB, nullptr, 4096, 1024, 1024);
  gemm_bias_kernel<0, true ><<<gMain, blk, 0, stream>>>(key_, Wk, bk, nullptr, nullptr,
                                              kB, nullptr, nullptr, 4096, 1024, 1024);
  gemm_bias_kernel<0, true ><<<gMain, blk, 0, stream>>>(value, Wv, bv, nullptr, nullptr,
                                              vB, nullptr, nullptr, 4096, 1024, 1024);
  gemm_bias_kernel<0, false><<<gPos, blk, 0, stream>>>(pos_emb, Wpos, nullptr, nullptr, nullptr,
                                              pB, nullptr, nullptr, 2047, 1024, 1024);

  // LDS: 131072(S) + 8192(q) + 32768(kt x2) + 24576(pt x2) + 12288(bdt) + 8192(vtT x2)
  attention_kernel<<<dim3(32, 16, 4), blk, 217088, stream>>>(quB, qvB, kB, vB, pB, mask, x);

  gemm_bias_kernel<2, true ><<<gMain, blk, 0, stream>>>(x, Wout, bout, nullptr, nullptr,
                                              nullptr, nullptr, (float*)d_out, 4096, 1024, 1024);
}
